// LNLSTM_11458972745817
// MI455X (gfx1250) — compile-verified, matrix operand order checked by reading, not run on hardware
//
#include <hip/hip_runtime.h>
#include <hip/hip_bf16.h>

// ---------------------------------------------------------------------------
// LN-LSTM for MI455X (gfx1250, wave32, WMMA).
// T=512, B=64, D=H=512, 4H=2048, L=2.
//
// Phase A (per layer, time-parallel): i2h = LN(X @ Wi^T + bi)  via bf16 WMMA.
// Phase B (per layer, sequential):    persistent single-WGP kernel, h/c live
//                                     in LDS, per-step 64x2048x512 bf16 WMMA
//                                     GEMM against L2-resident Wh.  Initial
//                                     cell state is DMA'd into LDS with the
//                                     Tensor Data Mover (tensor_load_to_lds).
// ---------------------------------------------------------------------------

#define TT   512
#define BB   64
#define DD   512
#define HH   512
#define G4   2048
#define LL   2
#define TB   (TT * BB)      // 32768

typedef __attribute__((ext_vector_type(16))) __bf16 v16bf;
typedef __attribute__((ext_vector_type(8)))  __bf16 v8bf;
typedef __attribute__((ext_vector_type(8)))  float  v8f;
typedef __attribute__((ext_vector_type(4)))  unsigned int v4u;
typedef __attribute__((ext_vector_type(8)))  unsigned int v8u;

union Frag16 { v16bf v; v8bf h[2]; };

// Load a 16-lane-rows x 32-K bf16 WMMA operand fragment.
// rowbase points at the first row of the 16-row tile; ldk is the K stride
// (elements); storage is row-major with K contiguous, so each lane's 16
// values are two contiguous 8-element (16B) runs:
//   k in [8*half, 8*half+8)  and  [16+8*half, 16+8*half+8),  half = lane>>4.
__device__ inline v16bf load_frag(const __bf16* rowbase, int ldk, int k0, int lane) {
    const __bf16* p = rowbase + (size_t)(lane & 15) * ldk + k0 + ((lane >> 4) << 3);
    Frag16 f;
    f.h[0] = *(const v8bf*)(p);
    f.h[1] = *(const v8bf*)(p + 16);
    return f.v;
}

__device__ inline v8f wmma_bf16(v16bf a, v16bf b, v8f c) {
    return __builtin_amdgcn_wmma_f32_16x16x32_bf16(
        /*neg_a=*/false, a, /*neg_b=*/false, b,
        /*c_mod=*/(short)0, c, /*reuse_a=*/false, /*reuse_b=*/false);
}

__device__ inline float sigm(float x) { return 1.f / (1.f + __expf(-x)); }

// ---------------------------------------------------------------------------
// f32 -> bf16 conversion (grid-stride)
// ---------------------------------------------------------------------------
__global__ void cvt_bf16(const float* __restrict__ in, __bf16* __restrict__ out, int n) {
    int i = blockIdx.x * blockDim.x + threadIdx.x;
    int stride = gridDim.x * blockDim.x;
    for (; i < n; i += stride) out[i] = (__bf16)in[i];
}

// ---------------------------------------------------------------------------
// Phase A GEMM: C[M x 2048] = A[M x 512](bf16) @ W[2048 x 512]^T (bf16)
// 256 threads = 8 waves; each wave computes a 16x64 strip (4 C tiles),
// reusing one A fragment across 4 WMMAs per K chunk.
// grid = (2048/64, M/128) = (32, 256)
// ---------------------------------------------------------------------------
__global__ __launch_bounds__(256)
void gemm_i2h(const __bf16* __restrict__ A, const __bf16* __restrict__ W,
              float* __restrict__ C) {
    const int lane = threadIdx.x & 31;
    const int w    = threadIdx.x >> 5;
    const int m0   = (blockIdx.y * 8 + w) * 16;
    const int n0   = blockIdx.x * 64;

    v8f acc[4] = {v8f{}, v8f{}, v8f{}, v8f{}};

    for (int k0 = 0; k0 < DD; k0 += 32) {
        v16bf a = load_frag(A + (size_t)m0 * DD, DD, k0, lane);
        __builtin_prefetch(W + (size_t)n0 * DD + k0 + 128, 0, 1);
#pragma unroll
        for (int nt = 0; nt < 4; ++nt) {
            v16bf b = load_frag(W + (size_t)(n0 + nt * 16) * DD, DD, k0, lane);
            acc[nt] = wmma_bf16(a, b, acc[nt]);
        }
    }

    // C/D layout: lane holds column n = lane&15 of the tile, VGPR r holds
    // row m = r + 8*(lane>>4).
    const int half = lane >> 4;
#pragma unroll
    for (int nt = 0; nt < 4; ++nt) {
        int n = n0 + nt * 16 + (lane & 15);
#pragma unroll
        for (int r = 0; r < 8; ++r) {
            C[(size_t)(m0 + r + 8 * half) * G4 + n] = acc[nt][r];
        }
    }
}

// ---------------------------------------------------------------------------
// Row LayerNorm over 2048 columns (unbiased variance, /2047), adds bias
// first, applies alpha/beta, in place.  One 256-thread block per row.
// ---------------------------------------------------------------------------
__global__ __launch_bounds__(256)
void ln_rows(float* __restrict__ X, const float* __restrict__ bias,
             const float* __restrict__ alpha, const float* __restrict__ beta) {
    __shared__ float s1[256], s2[256];
    float* xr = X + (size_t)blockIdx.x * G4;
    float v[8];
    float s = 0.f, q = 0.f;
#pragma unroll
    for (int i = 0; i < 8; ++i) {
        int c = threadIdx.x + i * 256;
        float t = xr[c] + bias[c];
        v[i] = t; s += t; q += t * t;
    }
    s1[threadIdx.x] = s; s2[threadIdx.x] = q;
    __syncthreads();
    for (int off = 128; off > 0; off >>= 1) {
        if (threadIdx.x < off) {
            s1[threadIdx.x] += s1[threadIdx.x + off];
            s2[threadIdx.x] += s2[threadIdx.x + off];
        }
        __syncthreads();
    }
    float mean = s1[0] * (1.f / 2048.f);
    float var  = (s2[0] - 2048.f * mean * mean) * (1.f / 2047.f);
    float rstd = rsqrtf(var + 1e-6f);
#pragma unroll
    for (int i = 0; i < 8; ++i) {
        int c = threadIdx.x + i * 256;
        xr[c] = alpha[c] * (v[i] - mean) * rstd + beta[c];
    }
}

// ---------------------------------------------------------------------------
// Phase B: persistent recurrence. One workgroup, 1024 threads = 32 waves.
// LDS: c (f32, 128KB) + h (bf16, 64KB) + o (bf16, 64KB) + reduction partials.
// Per step: 64x2048x512 bf16 WMMA GEMM (A = h in LDS, B = Wh in L2),
// gate LN over 2048, cell LN over 512, pointwise update, all on-WGP.
// ---------------------------------------------------------------------------
__global__ __launch_bounds__(1024)
void lstm_rec(const float* __restrict__ i2h,      // (T*B) x 2048, already LN'd
              const __bf16* __restrict__ Whb,     // 2048 x 512 bf16
              const float* __restrict__ bh,       // 2048
              const float* __restrict__ ah,       // 2048  (a_h2h)
              const float* __restrict__ bln,      // 2048  (b_h2h)
              const float* __restrict__ ac,       // 512   (a_cell)
              const float* __restrict__ bc,       // 512   (b_cell)
              const float* __restrict__ h0,       // 64 x 512
              const float* __restrict__ c0,       // 64 x 512
              float* __restrict__ G,              // 64 x 2048 scratch (L2)
              float* __restrict__ ys,             // (T*B) x 512 f32 or null
              __bf16* __restrict__ ysb) {         // (T*B) x 512 bf16 or null
    extern __shared__ char smem[];
    float*  c_lds  = (float*)smem;                          // 64*512 f32
    __bf16* h_lds  = (__bf16*)(smem + 131072);              // 64*512 bf16
    __bf16* o_lds  = (__bf16*)(smem + 131072 + 65536);      // 64*512 bf16
    float*  gpartS = (float*)(smem + 262144);               // [8][64]
    float*  gpartQ = gpartS + 512;                          // [8][64]
    float*  cpartS = gpartQ + 512;                          // [64][16]
    float*  cpartQ = cpartS + 1024;                         // [64][16]
    float*  gmean  = cpartQ + 1024;                         // [64]
    float*  grstd  = gmean + 64;
    float*  cmean  = grstd + 64;
    float*  crstd  = cmean + 64;

    const int tid  = threadIdx.x;
    const int lane = tid & 31;
    const int w    = tid >> 5;
    const int m4   = w & 3;        // which 16-row batch tile (B=64 -> 4 tiles)
    const int ng   = w >> 2;       // N-tile group 0..7
    const int m0   = m4 * 16;
    const int half = lane >> 4;

    // ---- Initial cell state via Tensor Data Mover: c0 (128 KB, f32) -> LDS.
    // D# per CDNA5 ISA ch.8: group0 = {count/flags, lds_addr, global_addr,
    // type=2}; group1 = {data_size=4B, tensor_dim0=tile_dim0=32768, dim1=1,
    // stride}; groups 2/3 describe (unused) higher dims.  Wave 0 issues the
    // DMA (TDM ignores EXEC; other waves branch around), waits TENSORcnt=0,
    // then the workgroup barrier publishes LDS to everyone.
    if (tid < 32) {
        unsigned long long ga = (unsigned long long)(const void*)c0;
        unsigned int lds_base = (unsigned int)__builtin_amdgcn_groupstaticsize();
        v4u g0 = { 1u,                                   // count=1, user D#
                   lds_base,                             // lds_addr (bytes)
                   (unsigned int)ga,                     // global_addr[31:0]
                   (unsigned int)((ga >> 32) & 0x01FFFFFFu) | (2u << 30) }; // [56:32] | type=2
        v8u g1 = { 0x00020000u,                          // data_size=2 (4B)
                   0x80000000u,                          // tensor_dim0 = 32768 (lo16 @ [31:16])
                   0x00010000u,                          // dim0 hi=0, tensor_dim1 = 1
                   0x80000000u,                          // dim1 hi=0, tile_dim0 = 32768
                   0x00000001u,                          // tile_dim1 = 1, tile_dim2 = 0
                   32768u,                               // tensor_dim0_stride lo
                   0u, 0u };
        v4u g2 = { 1u, 1u, 0u, 0u };                     // tensor_dim2=1, tensor_dim3=1
        v4u g3 = { 0u, 0u, 0u, 0u };
        asm volatile("tensor_load_to_lds %0, %1, %2, %3"
                     :: "s"(g0), "s"(g1), "s"(g2), "s"(g3)
                     : "memory");
        __builtin_amdgcn_s_wait_tensorcnt(0);
    }
    // Initial hidden state: f32 -> bf16 through VALU (TDM can't convert).
    for (int i = tid; i < BB * HH; i += 1024) {
        h_lds[i] = (__bf16)h0[i];
    }
    __syncthreads();

    for (int t = 0; t < TT; ++t) {
        // ---- 1) GEMM: G = h @ Wh^T + bh, with per-row sum / sumsq ----
        float rs[8] = {0, 0, 0, 0, 0, 0, 0, 0};
        float rq[8] = {0, 0, 0, 0, 0, 0, 0, 0};
        for (int i = 0; i < 16; ++i) {
            const int n0 = (ng + (i << 3)) * 16;   // tiles 0..127
            v8f acc = v8f{};
            for (int k0 = 0; k0 < HH; k0 += 32) {
                __builtin_prefetch(Whb + (size_t)n0 * HH + k0 + 128, 0, 1);
                v16bf a = load_frag(h_lds + (size_t)m0 * HH, HH, k0, lane);
                v16bf b = load_frag(Whb + (size_t)n0 * HH, HH, k0, lane);
                acc = wmma_bf16(a, b, acc);
            }
            const float bv = bh[n0 + (lane & 15)];
            const int   n  = n0 + (lane & 15);
#pragma unroll
            for (int r = 0; r < 8; ++r) {
                float val = acc[r] + bv;
                G[(size_t)(m0 + r + 8 * half) * G4 + n] = val;
                rs[r] += val;
                rq[r] += val * val;
            }
        }
        // fixed-order cross-lane reduce within each 16-lane half
#pragma unroll
        for (int r = 0; r < 8; ++r) {
#pragma unroll
            for (int mk = 1; mk < 16; mk <<= 1) {
                rs[r] += __shfl_xor(rs[r], mk, 32);
                rq[r] += __shfl_xor(rq[r], mk, 32);
            }
        }
        if ((lane & 15) == 0) {
            const int rowbase = m0 + 8 * half;
#pragma unroll
            for (int r = 0; r < 8; ++r) {
                gpartS[ng * 64 + rowbase + r] = rs[r];
                gpartQ[ng * 64 + rowbase + r] = rq[r];
            }
        }
        __threadfence();   // make G stores visible within the workgroup
        __syncthreads();

        // ---- 2) gate-LN stats per batch row (deterministic order) ----
        if (tid < BB) {
            float s = 0.f, q = 0.f;
#pragma unroll
            for (int p = 0; p < 8; ++p) { s += gpartS[p * 64 + tid]; q += gpartQ[p * 64 + tid]; }
            float mean = s * (1.f / 2048.f);
            float var  = (q - 2048.f * mean * mean) * (1.f / 2047.f);
            gmean[tid] = mean;
            grstd[tid] = rsqrtf(var + 1e-6f);
        }
        __syncthreads();

        // ---- 3) gates + cell pre-LN update ----
        {
            const int b  = tid >> 4;   // 0..63
            const int tj = tid & 15;
            const float mean = gmean[b], rstd = grstd[b];
            const float* Gb = G + (size_t)b * G4;
            const float* Ib = i2h + ((size_t)t * BB + b) * G4;
            float ls = 0.f, lq = 0.f;
            for (int jj = 0; jj < 32; ++jj) {
                const int j = tj + (jj << 4);
                float gi = ah[j]        * (Gb[j]        - mean) * rstd + bln[j]        + Ib[j];
                float gf = ah[j + 512]  * (Gb[j + 512]  - mean) * rstd + bln[j + 512]  + Ib[j + 512];
                float go = ah[j + 1024] * (Gb[j + 1024] - mean) * rstd + bln[j + 1024] + Ib[j + 1024];
                float gg = ah[j + 1536] * (Gb[j + 1536] - mean) * rstd + bln[j + 1536] + Ib[j + 1536];
                float it = sigm(gi), ft = sigm(gf), ot = sigm(go), gt = tanhf(gg);
                float cpre = c_lds[b * HH + j] * ft + it * gt;
                c_lds[b * HH + j] = cpre;
                o_lds[b * HH + j] = (__bf16)ot;
                ls += cpre; lq += cpre * cpre;
            }
            cpartS[b * 16 + tj] = ls;
            cpartQ[b * 16 + tj] = lq;
        }
        __syncthreads();

        // ---- 4) cell-LN stats ----
        if (tid < BB) {
            float s = 0.f, q = 0.f;
#pragma unroll
            for (int p = 0; p < 16; ++p) { s += cpartS[tid * 16 + p]; q += cpartQ[tid * 16 + p]; }
            float mean = s * (1.f / 512.f);
            float var  = (q - 512.f * mean * mean) * (1.f / 511.f);
            cmean[tid] = mean;
            crstd[tid] = rsqrtf(var + 1e-6f);
        }
        __syncthreads();

        // ---- 5) cell LN + h update ----
        {
            const int b  = tid >> 4;
            const int tj = tid & 15;
            const float mean = cmean[b], rstd = crstd[b];
            float*  ysrow = ys  ? ys  + ((size_t)t * BB + b) * HH : nullptr;
            __bf16* ybrow = ysb ? ysb + ((size_t)t * BB + b) * HH : nullptr;
            for (int jj = 0; jj < 32; ++jj) {
                const int j = tj + (jj << 4);
                float cn = ac[j] * (c_lds[b * HH + j] - mean) * rstd + bc[j];
                c_lds[b * HH + j] = cn;
                float hv = (float)o_lds[b * HH + j] * tanhf(cn);
                h_lds[b * HH + j] = (__bf16)hv;
                if (ysrow) ysrow[j] = hv;
                if (ybrow) ybrow[j] = (__bf16)hv;
            }
        }
        __syncthreads();
    }
}

// ---------------------------------------------------------------------------
// Host launcher
// ---------------------------------------------------------------------------
extern "C" void kernel_launch(void* const* d_in, const int* in_sizes, int n_in,
                              void* d_out, int out_size, void* d_ws, size_t ws_size,
                              hipStream_t stream) {
    const float* x      = (const float*)d_in[0];
    const float* h0     = (const float*)d_in[1];
    const float* c0     = (const float*)d_in[2];
    const float* Wi     = (const float*)d_in[3];
    const float* bi     = (const float*)d_in[4];
    const float* Wh     = (const float*)d_in[5];
    const float* bh     = (const float*)d_in[6];
    const float* a_i2h  = (const float*)d_in[7];
    const float* b_i2h  = (const float*)d_in[8];
    const float* a_h2h  = (const float*)d_in[9];
    const float* b_h2h  = (const float*)d_in[10];
    const float* a_cell = (const float*)d_in[11];
    const float* b_cell = (const float*)d_in[12];
    float* out = (float*)d_out;

    char*  ws  = (char*)d_ws;
    size_t off = 0;
    auto carve = [&](size_t bytes) -> void* {
        void* p = ws + off;
        off += (bytes + 255) & ~(size_t)255;
        return p;
    };
    __bf16* Xb   = (__bf16*)carve((size_t)TB * DD * 2);          // 32 MB
    __bf16* Wib  = (__bf16*)carve((size_t)LL * G4 * DD * 2);     //  4 MB
    __bf16* Whb  = (__bf16*)carve((size_t)LL * G4 * HH * 2);     //  4 MB
    float*  i2h  = (float*)carve((size_t)TB * G4 * 4);           // 256 MB
    __bf16* Ysb  = (__bf16*)carve((size_t)TB * HH * 2);          // 32 MB
    float*  Gsc  = (float*)carve((size_t)BB * G4 * 4);           // 512 KB

    cvt_bf16<<<2048, 256, 0, stream>>>(x,  Xb,  TB * DD);
    cvt_bf16<<<512,  256, 0, stream>>>(Wi, Wib, LL * G4 * DD);
    cvt_bf16<<<512,  256, 0, stream>>>(Wh, Whb, LL * G4 * HH);

    const size_t rec_smem = 131072 + 65536 + 65536 + 13312;      // 275,456 B < 320 KB

    for (int l = 0; l < LL; ++l) {
        const __bf16* Ain = (l == 0) ? Xb : Ysb;
        gemm_i2h<<<dim3(G4 / 64, TB / 128), 256, 0, stream>>>(
            Ain, Wib + (size_t)l * G4 * DD, i2h);
        ln_rows<<<TB, 256, 0, stream>>>(
            i2h, bi + (size_t)l * G4, a_i2h + (size_t)l * G4, b_i2h + (size_t)l * G4);
        lstm_rec<<<1, 1024, rec_smem, stream>>>(
            i2h, Whb + (size_t)l * G4 * HH, bh + (size_t)l * G4,
            a_h2h + (size_t)l * G4, b_h2h + (size_t)l * G4,
            a_cell + (size_t)l * HH, b_cell + (size_t)l * HH,
            h0 + (size_t)l * BB * HH, c0 + (size_t)l * BB * HH,
            Gsc,
            (l == LL - 1) ? out : nullptr,
            (l == 0) ? Ysb : nullptr);
    }
    (void)in_sizes; (void)n_in; (void)out_size; (void)ws_size;
}
